// MultiHeadAttention_63307817943761
// MI455X (gfx1250) — matrix operand order repaired and checked by reading, not compiled, not run on hardware
//
#include <hip/hip_runtime.h>
#include <hip/hip_bf16.h>

#define B 2
#define S 2048
#define D 1024
#define H 16
#define DK 64
#define DV 64

typedef __attribute__((ext_vector_type(16))) _Float16 v16h;
typedef __attribute__((ext_vector_type(8)))  _Float16 v8h;
typedef __attribute__((ext_vector_type(8)))  float    v8f;
typedef __attribute__((ext_vector_type(4)))  float    vf4;

union F16x16 { v16h v; v8h h2[2]; _Float16 e[16]; };

// ---- WMMA fragment loaders (CDNA5 16x16x32 f16 layouts, wave32) ----

// A 16x32 f16: lane%16 = row M; lanes 0-15 hold K=0..7 (halves0-7) and K=16..23
// (halves8-15); lanes 16-31 hold K=8..15 and K=24..31.
static __device__ inline v16h wmma_load_a_f16(const _Float16* tile, int lda, int lane) {
  const int row = lane & 15;
  const int kb  = (lane >> 4) << 3;           // 0 or 8
  const _Float16* p = tile + row * lda + kb;
  F16x16 u;
  u.h2[0] = *(const v8h*)(p);                 // K = kb .. kb+7
  u.h2[1] = *(const v8h*)(p + 16);            // K = kb+16 .. kb+23
  return u.v;
}

// Same A layout, fp32 source, NON-TEMPORAL loads (streaming attention reads).
static __device__ inline v16h wmma_load_a_f32_nt(const float* tile, int lda, int lane) {
  const int row = lane & 15;
  const int kb  = (lane >> 4) << 3;
  const float* p = tile + row * lda + kb;
  vf4 f0 = __builtin_nontemporal_load((const vf4*)(p));
  vf4 f1 = __builtin_nontemporal_load((const vf4*)(p + 4));
  vf4 f2 = __builtin_nontemporal_load((const vf4*)(p + 16));
  vf4 f3 = __builtin_nontemporal_load((const vf4*)(p + 20));
  F16x16 u;
#pragma unroll
  for (int i = 0; i < 4; ++i) {
    u.e[i]      = (_Float16)f0[i];
    u.e[4 + i]  = (_Float16)f1[i];
    u.e[8 + i]  = (_Float16)f2[i];
    u.e[12 + i] = (_Float16)f3[i];
  }
  return u.v;
}

// Same A layout, fp32 source, regular (L2-resident activations).
static __device__ inline v16h wmma_load_a_f32(const float* tile, int lda, int lane) {
  const int row = lane & 15;
  const int kb  = (lane >> 4) << 3;
  const float* p = tile + row * lda + kb;
  vf4 f0 = *(const vf4*)(p);
  vf4 f1 = *(const vf4*)(p + 4);
  vf4 f2 = *(const vf4*)(p + 16);
  vf4 f3 = *(const vf4*)(p + 20);
  F16x16 u;
#pragma unroll
  for (int i = 0; i < 4; ++i) {
    u.e[i]      = (_Float16)f0[i];
    u.e[4 + i]  = (_Float16)f1[i];
    u.e[8 + i]  = (_Float16)f2[i];
    u.e[12 + i] = (_Float16)f3[i];
  }
  return u.v;
}

// B 32x16 f16 from a COLUMN-CONTIGUOUS (transposed) source bt[n*ldb + k]:
// lane%16 = column N; lanes 0-15 hold K=0..7,16..23, lanes 16-31 hold K=8..15,24..31.
static __device__ inline v16h wmma_load_bt_f16(const _Float16* bt, int ldb, int lane) {
  const int col = lane & 15;
  const int kb  = (lane >> 4) << 3;           // 0 or 8
  const _Float16* p = bt + col * ldb + kb;
  F16x16 u;
  u.h2[0] = *(const v8h*)(p);
  u.h2[1] = *(const v8h*)(p + 16);
  return u.v;
}

static __device__ inline v8f wmma32(v16h a, v16h b, v8f c) {
  return __builtin_amdgcn_wmma_f32_16x16x32_f16(false, a, false, b, (short)0, c, false, false);
}

// ---- Kernel 0: fp32 [K=1024, N=1024] -> f16 transposed [N,K] ----
__global__ __launch_bounds__(256) void k_cvt_t(const float* __restrict__ W,
                                               _Float16* __restrict__ Wt) {
  int idx = blockIdx.x * 256 + threadIdx.x;   // 0 .. 1024*1024-1
  int k = idx >> 10, n = idx & 1023;
  Wt[n * 1024 + k] = (_Float16)W[idx];
}

// ---- Kernel 1: projection GEMM [4096,1024]x[1024,1024]+bias -> f16 heads ----
// mode 0: Q -> [B,H,S,DK] scaled by 1/sqrt(DK)
// mode 1: K -> [B,H,S,DK]
// mode 2: V -> transposed [B,H,DV,S]
__global__ __launch_bounds__(256) void k_proj(const float* __restrict__ X,
                                              const _Float16* __restrict__ Wt,
                                              const float* __restrict__ bias,
                                              _Float16* __restrict__ dst, int mode) {
  const int lane = threadIdx.x & 31;
  const int wid  = blockIdx.x * 8 + (threadIdx.x >> 5);   // 0..4095
  const int m0 = (wid >> 4) << 4;                          // 256 row tiles
  const int n0 = (wid & 15) << 6;                          // 16 col groups of 64
  v8f c[4];
#pragma unroll
  for (int j = 0; j < 4; ++j)
#pragma unroll
    for (int r = 0; r < 8; ++r) c[j][r] = 0.0f;

  const float* Atile = X + (size_t)m0 * D;

  // ping-pong double buffer: no register rotation, loads land in the idle set
  v16h a0 = wmma_load_a_f32(Atile, D, lane);
  v16h b0[4];
#pragma unroll
  for (int j = 0; j < 4; ++j)
    b0[j] = wmma_load_bt_f16(Wt + (size_t)(n0 + j * 16) * D, D, lane);

  for (int kk = 0; kk < D; kk += 64) {
    v16h a1 = wmma_load_a_f32(Atile + kk + 32, D, lane);   // always in-bounds
    v16h b1[4];
#pragma unroll
    for (int j = 0; j < 4; ++j)
      b1[j] = wmma_load_bt_f16(Wt + (size_t)(n0 + j * 16) * D + kk + 32, D, lane);
#pragma unroll
    for (int j = 0; j < 4; ++j) c[j] = wmma32(a0, b0[j], c[j]);
    if (kk + 64 < D) {
      a0 = wmma_load_a_f32(Atile + kk + 64, D, lane);
#pragma unroll
      for (int j = 0; j < 4; ++j)
        b0[j] = wmma_load_bt_f16(Wt + (size_t)(n0 + j * 16) * D + kk + 64, D, lane);
    }
#pragma unroll
    for (int j = 0; j < 4; ++j) c[j] = wmma32(a1, b1[j], c[j]);
  }

  const int mB = (lane >> 4) << 3;
  const int nc = lane & 15;
#pragma unroll
  for (int j = 0; j < 4; ++j)
#pragma unroll
    for (int r = 0; r < 8; ++r) {
      int m = m0 + mB + r;
      int n = n0 + j * 16 + nc;
      float v = c[j][r] + bias[n];
      int b = m >> 11, s = m & (S - 1);
      int h = n >> 6, d = n & 63;
      if (mode == 2) {
        dst[((size_t)((b * H + h) * DV + d)) * S + s] = (_Float16)v;
      } else {
        if (mode == 0) v *= 0.125f;                       // 1/sqrt(64)
        dst[((size_t)((b * H + h) * S + s)) * DK + d] = (_Float16)v;
      }
    }
}

// ---- Kernel 2: FUSED energy + mask + row softmax -> single attn store ----
// One block owns a 16-row x 2048-col strip of one (b,h). 8 waves x 256 cols,
// 16 accumulator tiles per wave (scores stay in VGPRs end-to-end).
__global__ __launch_bounds__(256) void k_scores_softmax(const _Float16* __restrict__ Qh,
                                                        const _Float16* __restrict__ Kh,
                                                        const int* __restrict__ mask,
                                                        float* __restrict__ attn) {
  const int lane  = threadIdx.x & 31;
  const int w     = threadIdx.x >> 5;        // wave 0..7
  const int mt    = blockIdx.x & 127;
  const int bh    = blockIdx.x >> 7;         // 0..31
  const int m0    = mt << 4;
  const int nbase = w << 8;                  // this wave's 256-col slice

  const _Float16* Atile = Qh + ((size_t)bh * S + m0) * DK;
  const _Float16* Btile = Kh + ((size_t)bh * S) * DK;

  v8f c[16];
#pragma unroll
  for (int j = 0; j < 16; ++j)
#pragma unroll
    for (int r = 0; r < 8; ++r) c[j][r] = 0.0f;

#pragma unroll
  for (int kk = 0; kk < DK; kk += 32) {      // 2 steps
    v16h a = wmma_load_a_f16(Atile + kk, DK, lane);
#pragma unroll
    for (int j = 0; j < 16; ++j) {
      v16h b = wmma_load_bt_f16(Btile + (size_t)(nbase + j * 16) * DK + kk, DK, lane);
      c[j] = wmma32(a, b, c[j]);
    }
  }

  const int bb = bh >> 4;
  const int mB = (lane >> 4) << 3;           // rows 0..7 or 8..15
  const int nc = lane & 15;

  // mask + per-lane row max
  float rmax[8];
#pragma unroll
  for (int r = 0; r < 8; ++r) rmax[r] = -3.0e38f;
#pragma unroll
  for (int j = 0; j < 16; ++j) {
    const int n = nbase + j * 16 + nc;
#pragma unroll
    for (int r = 0; r < 8; ++r) {
      const int m = m0 + mB + r;
      int mv = mask[((size_t)bb * S + m) * S + n];
      float e = mv ? c[j][r] : -10000.0f;
      c[j][r] = e;
      rmax[r] = fmaxf(rmax[r], e);
    }
  }
  // reduce across the 16 lanes of each half-wave (rows differ per half)
#pragma unroll
  for (int off = 8; off >= 1; off >>= 1)
#pragma unroll
    for (int r = 0; r < 8; ++r) rmax[r] = fmaxf(rmax[r], __shfl_xor(rmax[r], off, 32));

  __shared__ float wred[8][16];
  __shared__ float wsum[8][16];
  if (nc == 0) {
#pragma unroll
    for (int r = 0; r < 8; ++r) wred[w][mB + r] = rmax[r];
  }
  __syncthreads();
  float gmax[8];
#pragma unroll
  for (int r = 0; r < 8; ++r) {
    float g = wred[0][mB + r];
#pragma unroll
    for (int ww = 1; ww < 8; ++ww) g = fmaxf(g, wred[ww][mB + r]);
    gmax[r] = g;
  }
  // exp + per-lane row sum
  float rsum[8];
#pragma unroll
  for (int r = 0; r < 8; ++r) rsum[r] = 0.0f;
#pragma unroll
  for (int j = 0; j < 16; ++j)
#pragma unroll
    for (int r = 0; r < 8; ++r) {
      float e = __expf(c[j][r] - gmax[r]);
      c[j][r] = e;
      rsum[r] += e;
    }
#pragma unroll
  for (int off = 8; off >= 1; off >>= 1)
#pragma unroll
    for (int r = 0; r < 8; ++r) rsum[r] += __shfl_xor(rsum[r], off, 32);
  if (nc == 0) {
#pragma unroll
    for (int r = 0; r < 8; ++r) wsum[w][mB + r] = rsum[r];
  }
  __syncthreads();
  float ginv[8];
#pragma unroll
  for (int r = 0; r < 8; ++r) {
    float s = wsum[0][mB + r];
#pragma unroll
    for (int ww = 1; ww < 8; ++ww) s += wsum[ww][mB + r];
    ginv[r] = 1.0f / s;
  }
  // single normalized store, streaming (bypass L2 retention)
#pragma unroll
  for (int j = 0; j < 16; ++j) {
    const int n = nbase + j * 16 + nc;
#pragma unroll
    for (int r = 0; r < 8; ++r) {
      const int m = m0 + mB + r;
      __builtin_nontemporal_store(c[j][r] * ginv[r],
                                  &attn[((size_t)bh * S + m) * S + n]);
    }
  }
}

// ---- Kernel 3: ctx = attn @ V (streaming fp32 A, Vt column-major) ----
__global__ __launch_bounds__(256) void k_av(const float* __restrict__ attn,
                                            const _Float16* __restrict__ Vt,
                                            _Float16* __restrict__ ctx) {
  const int lane = threadIdx.x & 31;
  const int wid  = blockIdx.x * 8 + (threadIdx.x >> 5);   // 0..4095
  const int m0 = (wid & 127) << 4;
  const int bh = wid >> 7;
  const float* Atile = attn + ((size_t)bh * S + m0) * S;
  const _Float16* Btile = Vt + (size_t)bh * DV * S;
  v8f c[4];
#pragma unroll
  for (int j = 0; j < 4; ++j)
#pragma unroll
    for (int r = 0; r < 8; ++r) c[j][r] = 0.0f;

  v16h a0 = wmma_load_a_f32_nt(Atile, S, lane);
  v16h b0[4];
#pragma unroll
  for (int j = 0; j < 4; ++j)
    b0[j] = wmma_load_bt_f16(Btile + (size_t)(j * 16) * S, S, lane);

  for (int kk = 0; kk < S; kk += 64) {                    // 32 double-steps
    v16h a1 = wmma_load_a_f32_nt(Atile + kk + 32, S, lane);
    v16h b1[4];
#pragma unroll
    for (int j = 0; j < 4; ++j)
      b1[j] = wmma_load_bt_f16(Btile + (size_t)(j * 16) * S + kk + 32, S, lane);
#pragma unroll
    for (int j = 0; j < 4; ++j) c[j] = wmma32(a0, b0[j], c[j]);
    if (kk + 64 < S) {
      a0 = wmma_load_a_f32_nt(Atile + kk + 64, S, lane);
#pragma unroll
      for (int j = 0; j < 4; ++j)
        b0[j] = wmma_load_bt_f16(Btile + (size_t)(j * 16) * S + kk + 64, S, lane);
    }
#pragma unroll
    for (int j = 0; j < 4; ++j) c[j] = wmma32(a1, b1[j], c[j]);
  }
  const int bb = bh >> 4, h = bh & 15;
  const int mB = (lane >> 4) << 3;
  const int nc = lane & 15;
#pragma unroll
  for (int j = 0; j < 4; ++j)
#pragma unroll
    for (int r = 0; r < 8; ++r) {
      int s = m0 + mB + r;
      int dv = j * 16 + nc;
      ctx[((size_t)(bb * S + s)) * (H * DV) + h * DV + dv] = (_Float16)c[j][r];
    }
}

// ---- Kernel 4: out = ctx @ Wo + bo (fp32 out [B,S,D]) ----
__global__ __launch_bounds__(256) void k_out(const _Float16* __restrict__ ctx,
                                             const _Float16* __restrict__ Wot,
                                             const float* __restrict__ bo,
                                             float* __restrict__ out) {
  const int lane = threadIdx.x & 31;
  const int wid  = blockIdx.x * 8 + (threadIdx.x >> 5);
  const int m0 = (wid >> 4) << 4;
  const int n0 = (wid & 15) << 6;
  v8f c[4];
#pragma unroll
  for (int j = 0; j < 4; ++j)
#pragma unroll
    for (int r = 0; r < 8; ++r) c[j][r] = 0.0f;

  const _Float16* Atile = ctx + (size_t)m0 * D;
  v16h a0 = wmma_load_a_f16(Atile, D, lane);
  v16h b0[4];
#pragma unroll
  for (int j = 0; j < 4; ++j)
    b0[j] = wmma_load_bt_f16(Wot + (size_t)(n0 + j * 16) * D, D, lane);

  for (int kk = 0; kk < D; kk += 64) {
    v16h a1 = wmma_load_a_f16(Atile + kk + 32, D, lane);
    v16h b1[4];
#pragma unroll
    for (int j = 0; j < 4; ++j)
      b1[j] = wmma_load_bt_f16(Wot + (size_t)(n0 + j * 16) * D + kk + 32, D, lane);
#pragma unroll
    for (int j = 0; j < 4; ++j) c[j] = wmma32(a0, b0[j], c[j]);
    if (kk + 64 < D) {
      a0 = wmma_load_a_f16(Atile + kk + 64, D, lane);
#pragma unroll
      for (int j = 0; j < 4; ++j)
        b0[j] = wmma_load_bt_f16(Wot + (size_t)(n0 + j * 16) * D + kk + 64, D, lane);
    }
#pragma unroll
    for (int j = 0; j < 4; ++j) c[j] = wmma32(a1, b1[j], c[j]);
  }
  const int mB = (lane >> 4) << 3;
  const int nc = lane & 15;
#pragma unroll
  for (int j = 0; j < 4; ++j)
#pragma unroll
    for (int r = 0; r < 8; ++r) {
      int m = m0 + mB + r;
      int n = n0 + j * 16 + nc;
      out[(size_t)m * D + n] = c[j][r] + bo[n];
    }
}

extern "C" void kernel_launch(void* const* d_in, const int* in_sizes, int n_in,
                              void* d_out, int out_size, void* d_ws, size_t ws_size,
                              hipStream_t stream) {
  const float* query = (const float*)d_in[0];
  const float* key   = (const float*)d_in[1];
  const float* value = (const float*)d_in[2];
  const int*   mask  = (const int*)d_in[3];
  const float* Wq = (const float*)d_in[4];
  const float* bq = (const float*)d_in[5];
  const float* Wk = (const float*)d_in[6];
  const float* bk = (const float*)d_in[7];
  const float* Wv = (const float*)d_in[8];
  const float* bv = (const float*)d_in[9];
  const float* Wo = (const float*)d_in[10];
  const float* bo = (const float*)d_in[11];

  float* out  = (float*)d_out;                 // [B,S,D]
  float* attn = out + (size_t)B * S * D;       // [B,H,S,S]

  // f16 workspace staging (40 MB total)
  _Float16* Wqt = (_Float16*)d_ws;
  _Float16* Wkt = Wqt + (size_t)1024 * 1024;
  _Float16* Wvt = Wkt + (size_t)1024 * 1024;
  _Float16* Wot = Wvt + (size_t)1024 * 1024;
  _Float16* Qh  = Wot + (size_t)1024 * 1024;   // [B,H,S,DK]
  _Float16* Kh  = Qh  + (size_t)B * H * S * DK;
  _Float16* Vt  = Kh  + (size_t)B * H * S * DK; // [B,H,DV,S]
  _Float16* ctx = Vt  + (size_t)B * H * DV * S; // [B*S, H*DV]

  dim3 blk(256);
  k_cvt_t<<<4096, blk, 0, stream>>>(Wq, Wqt);
  k_cvt_t<<<4096, blk, 0, stream>>>(Wk, Wkt);
  k_cvt_t<<<4096, blk, 0, stream>>>(Wv, Wvt);
  k_cvt_t<<<4096, blk, 0, stream>>>(Wo, Wot);

  k_proj<<<512, blk, 0, stream>>>(query, Wqt, bq, Qh, 0);
  k_proj<<<512, blk, 0, stream>>>(key,   Wkt, bk, Kh, 1);
  k_proj<<<512, blk, 0, stream>>>(value, Wvt, bv, Vt, 2);

  k_scores_softmax<<<4096, blk, 0, stream>>>(Qh, Kh, mask, attn);
  k_av<<<512, blk, 0, stream>>>(attn, Vt, ctx);
  k_out<<<512, blk, 0, stream>>>(ctx, Wot, bo, out);
}
